// Encoder_37340445671714
// MI455X (gfx1250) — matrix operand order repaired and checked by reading, not compiled, not run on hardware
//
#include <hip/hip_runtime.h>

// ---------------------------------------------------------------------------
// MI455X (gfx1250) video transformer encoder, bf16 WMMA + TDM staging.
// B=4 L=8 D=512 NH=8 DK=64 FF=2048 NL=6 NP=196 S=1576, M=B*S=6304.
// Workspace use: ~168 MB (bf16 weights + fp32/bf16 activations).
// ---------------------------------------------------------------------------

typedef __bf16 bf16_t;
typedef __attribute__((ext_vector_type(16))) __bf16 v16bf;
typedef __attribute__((ext_vector_type(8)))  float  v8f;
typedef __attribute__((ext_vector_type(4)))  unsigned int u32x4;
typedef __attribute__((ext_vector_type(4)))  int i32x4;
typedef __attribute__((ext_vector_type(8)))  int i32x8;

#define CB  4
#define CL  8
#define CD  512
#define CNH 8
#define CDK 64
#define CFF 2048
#define CNL 6
#define CNP 196
#define CS  1576          // L*(NP+1)
#define CM  (CB*CS)       // 6304
#define PD  768           // 16*16*3

union FragB16 {
    v16bf v;
    unsigned int   u[8];
    unsigned short h[16];
};

__device__ __forceinline__ bf16_t f2bf(float x) {
    union { float f; unsigned int u; } c; c.f = x;
    unsigned int r = c.u + 0x7FFFu + ((c.u >> 16) & 1u);   // RNE
    union { unsigned short s; bf16_t b; } o; o.s = (unsigned short)(r >> 16);
    return o.b;
}

// A-fragment, 16x32 bf16 (contiguous k order, same as the B loaders): lane = half*16 + M; VGPR v holds the
// K pair half*16 + 2v.  rowptr already points at (row, k0).
__device__ __forceinline__ v16bf load_a_frag(const bf16_t* rowptr, int half) {
    FragB16 f;
#pragma unroll
    for (int vi = 0; vi < 8; ++vi) {
        int koff = (half << 4) + (vi << 1);
        f.u[vi] = *(const unsigned int*)(rowptr + koff);
    }
    return f.v;
}

// B-fragment, 32x16 bf16, where B = SrcT (Src stored row-major [rows][K]).
// lane = khalf*16 + n ; VGPR v holds K pair khalf*16 + 2v.
__device__ __forceinline__ v16bf load_bT_frag(const bf16_t* base, int ld, int lane) {
    FragB16 f;
    int n = lane & 15, kh = lane >> 4;
#pragma unroll
    for (int vi = 0; vi < 8; ++vi)
        f.u[vi] = *(const unsigned int*)(base + (long)n * ld + kh * 16 + vi * 2);
    return f.v;
}

// B-fragment with element (k,n) = Src[k][ncol+n] (row stride ld) -- for P.V.
__device__ __forceinline__ v16bf load_b_frag_kmaj(const bf16_t* base, int ld,
                                                  int ncol, int lane) {
    FragB16 f;
    int n = lane & 15, kh = lane >> 4;
#pragma unroll
    for (int vi = 0; vi < 8; ++vi) {
        const int k0 = kh * 16 + vi * 2;
        f.h[2 * vi]     = *(const unsigned short*)(base + (long)k0 * ld + ncol + n);
        f.h[2 * vi + 1] = *(const unsigned short*)(base + (long)(k0 + 1) * ld + ncol + n);
    }
    return f.v;
}

// ---------------------------------------------------------------------------
// TDM: 2D tile load Global->LDS (ISA ch.8 D# layout).  data_size = 2 bytes.
// Tile = tile_d1 rows x tile_d0 elements; row stride stride_elems; LDS rows
// padded by pad codes (interval/amount per D# encoding).
// ---------------------------------------------------------------------------
__device__ __forceinline__ void tdm_load_2d(unsigned lds_off, const void* gaddr,
                                            unsigned tensor_d0, unsigned tensor_d1,
                                            unsigned tile_d0, unsigned tile_d1,
                                            unsigned stride_elems,
                                            unsigned pad_int_code, unsigned pad_amt_code) {
    const unsigned long long ga = (unsigned long long)gaddr;
    u32x4 g0;
    g0[0] = 1u;                                              // count=1, user D#
    g0[1] = lds_off;                                         // lds_addr (bytes)
    g0[2] = (unsigned)(ga & 0xFFFFFFFFu);                    // global_addr[31:0]
    g0[3] = (unsigned)((ga >> 32) & 0x01FFFFFFu) | (2u << 30); // addr[56:32] | type=2
    i32x8 g1;
    g1[0] = (int)((1u << 16)                                 // data_size = 2B
                  | (1u << 20)                               // pad_enable
                  | (pad_int_code << 22) | (pad_amt_code << 25));
    g1[1] = (int)((tensor_d0 & 0xFFFFu) << 16);              // tensor_dim0[15:0]
    g1[2] = (int)(((tensor_d0 >> 16) & 0xFFFFu) | ((tensor_d1 & 0xFFFFu) << 16));
    g1[3] = (int)(((tensor_d1 >> 16) & 0xFFFFu) | ((tile_d0 & 0xFFFFu) << 16));
    g1[4] = (int)(tile_d1 & 0xFFFFu);                        // tile_dim1 ; tile_dim2=0
    g1[5] = (int)stride_elems;                               // tensor_dim0_stride[31:0]
    g1[6] = 0;
    g1[7] = 0;
    i32x4 z4 = {0, 0, 0, 0};
#if __clang_major__ >= 23
    i32x8 z8 = {0, 0, 0, 0, 0, 0, 0, 0};
    __builtin_amdgcn_tensor_load_to_lds(g0, g1, z4, z4, z8, 0);
#else
    __builtin_amdgcn_tensor_load_to_lds(g0, g1, z4, z4, 0);
#endif
}

// ---------------------------------------------------------------------------
// Generic bf16 WMMA GEMM: C[M,N] = A[M,K] * W[N,K]^T + bias (+residual)(+relu)
// 256 threads = 8 waves; block tile 128x64; wave tile 16(M) x 64(N).
// Weight tile (64 rows x 32 K, bf16) staged by the TDM, double-buffered.
// ---------------------------------------------------------------------------
#define LDSW 40                 // padded LDS row stride: 16 dwords row + 4 pad
#define WBUFB (64 * LDSW * 2)   // bytes per LDS weight buffer (5120)

__global__ void gemm_bf16_wmma(const bf16_t* __restrict__ A,
                               const bf16_t* __restrict__ W,
                               const float*  __restrict__ bias,
                               const float*  __restrict__ residual,
                               float*        __restrict__ outF,
                               bf16_t*       __restrict__ outB,
                               int M, int N, int K, int relu) {
    // Only __shared__ object in this kernel -> LDS offset 0 (TDM lds_addr base).
    __shared__ __align__(16) bf16_t ldsW[2][64 * LDSW];
    const int tid = threadIdx.x, lane = tid & 31, wave = tid >> 5;
    const int mb = blockIdx.x * 128, nb = blockIdx.y * 64;
    const int mrow0 = mb + wave * 16;
    const int half = lane >> 4;

    // The TDM writes LDS behind the compiler's back (descriptor-only builtin):
    // escape the buffer so its loads cannot be folded to undef.
    bf16_t* ldsw0 = &ldsW[0][0];
    asm volatile("" :: "r"(ldsw0) : "memory");

    v8f acc[4] = {};

    int arow = mrow0 + (lane & 15);
    if (arow >= M) arow = M - 1;
    const bf16_t* arowp = A + (long)arow * K;

    // prologue: TDM pull of the first 64x32 weight tile into buffer 0
    if (wave == 0) {
        tdm_load_2d(0u, W + (long)nb * K, (unsigned)K, 1u << 20,
                    32u, 64u, (unsigned)K, 3u /*16 dw*/, 3u /*4 dw pad*/);
        __builtin_amdgcn_s_wait_tensorcnt(0);
    }
    __syncthreads();

    int cur = 0;
    for (int k0 = 0; k0 < K; k0 += 32) {
        // prefetch next tile into the other buffer while we compute
        if (wave == 0 && k0 + 32 < K) {
            tdm_load_2d((unsigned)((cur ^ 1) * WBUFB),
                        W + (long)nb * K + (k0 + 32), (unsigned)K, 1u << 20,
                        32u, 64u, (unsigned)K, 3u, 3u);
        }
        __builtin_prefetch(arowp + k0 + 64, 0, 3);           // global_prefetch_b8

        asm volatile("" ::: "memory");   // TDM landed: force fresh LDS reads

        const v16bf af = load_a_frag(arowp + k0, half);
        const bf16_t* wb = ldsW[cur];
#pragma unroll
        for (int nt = 0; nt < 4; ++nt) {
            v16bf bf = load_bT_frag(wb + nt * 16 * LDSW, LDSW, lane);
            acc[nt] = __builtin_amdgcn_wmma_f32_16x16x32_bf16(
                false, af, false, bf, (short)0, acc[nt], false, false);
        }

        if (wave == 0) __builtin_amdgcn_s_wait_tensorcnt(0); // next buffer landed
        __syncthreads();                                     // reads of cur done too
        cur ^= 1;
    }

    // epilogue
#pragma unroll
    for (int nt = 0; nt < 4; ++nt) {
        const int col = nb + nt * 16 + (lane & 15);
        const float bv = bias ? bias[col] : 0.f;
#pragma unroll
        for (int r = 0; r < 8; ++r) {
            const int row = mrow0 + r + (half << 3);
            if (row < M) {
                float v = acc[nt][r] + bv;
                if (residual) v += residual[(long)row * N + col];
                if (relu) v = fmaxf(v, 0.f);
                if (outF) outF[(long)row * N + col] = v;
                if (outB) outB[(long)row * N + col] = f2bf(v);
            }
        }
    }
}

// ---------------------------------------------------------------------------
// Flash attention: one workgroup (4 waves) per (64 query rows, head, batch).
// K/V 32x64 tiles staged once per block into LDS; P tile reshaped via LDS.
// ---------------------------------------------------------------------------
#define ALD 80   // padded LDS row stride for K/V tiles (elements)

__global__ void attention_wmma(const bf16_t* __restrict__ Q,
                               const bf16_t* __restrict__ Kb,
                               const bf16_t* __restrict__ V,
                               bf16_t*       __restrict__ O) {
    __shared__ __align__(16) bf16_t lsK[32 * ALD];
    __shared__ __align__(16) bf16_t lsV[32 * ALD];
    __shared__ __align__(16) bf16_t ldsP[4][16 * 32];
    const int tid = threadIdx.x, lane = tid & 31, wave = tid >> 5;
    const int half = lane >> 4;
    const int b = blockIdx.z, h = blockIdx.y;
    const int q0 = blockIdx.x * 64 + wave * 16;

    const bf16_t* qb = Q + ((long)b * CS) * CD + h * CDK;
    const bf16_t* kb = Kb + ((long)b * CS) * CD + h * CDK;
    const bf16_t* vb = V + ((long)b * CS) * CD + h * CDK;

    int qrow = q0 + (lane & 15);
    if (qrow >= CS) qrow = CS - 1;
    const v16bf qf0 = load_a_frag(qb + (long)qrow * CD, half);
    const v16bf qf1 = load_a_frag(qb + (long)qrow * CD + 32, half);

    v8f oacc[4] = {};
    float mrun[8], srun[8];
#pragma unroll
    for (int r = 0; r < 8; ++r) { mrun[r] = -1e30f; srun[r] = 0.f; }

    const float scl = 0.125f;   // 1/sqrt(DK)

    for (int t0 = 0; t0 < CS; t0 += 32) {
        // ---- stage K and V tiles (32 rows x 64 elems, clamped at S tail) ----
        {
            const int r = tid >> 2, seg = tid & 3;           // 32 rows x 4 segs
            int srow = t0 + r;
            if (srow >= CS) srow = CS - 1;
            *(uint4*)(lsK + r * ALD + seg * 16) =
                *(const uint4*)(kb + (long)srow * CD + seg * 16);
            *(uint4*)(lsV + r * ALD + seg * 16) =
                *(const uint4*)(vb + (long)srow * CD + seg * 16);
        }
        __syncthreads();

        // ---- scores: Q(16x64) . K^T(64x32) -> two 16x16 tiles ----
        const bool msk0 = (t0 + (lane & 15)) >= CS;
        const bool msk1 = (t0 + 16 + (lane & 15)) >= CS;

        v8f s0 = {}, s1 = {};
        {
            v16bf kf = load_bT_frag(lsK, ALD, lane);
            s0 = __builtin_amdgcn_wmma_f32_16x16x32_bf16(false, qf0, false, kf, (short)0, s0, false, false);
            kf = load_bT_frag(lsK + 32, ALD, lane);
            s0 = __builtin_amdgcn_wmma_f32_16x16x32_bf16(false, qf1, false, kf, (short)0, s0, false, false);
            kf = load_bT_frag(lsK + 16 * ALD, ALD, lane);
            s1 = __builtin_amdgcn_wmma_f32_16x16x32_bf16(false, qf0, false, kf, (short)0, s1, false, false);
            kf = load_bT_frag(lsK + 16 * ALD + 32, ALD, lane);
            s1 = __builtin_amdgcn_wmma_f32_16x16x32_bf16(false, qf1, false, kf, (short)0, s1, false, false);
        }

        // ---- online softmax over this 32-column slab ----
        float corr[8];
#pragma unroll
        for (int r = 0; r < 8; ++r) {
            float a0 = msk0 ? -1e30f : s0[r] * scl;
            float a1 = msk1 ? -1e30f : s1[r] * scl;
            float mx = fmaxf(a0, a1);
#pragma unroll
            for (int m = 1; m < 16; m <<= 1) mx = fmaxf(mx, __shfl_xor(mx, m, 32));
            const float mnew = fmaxf(mrun[r], mx);
            corr[r] = __expf(mrun[r] - mnew);
            mrun[r] = mnew;
            const float p0 = __expf(a0 - mnew);
            const float p1 = __expf(a1 - mnew);
            float ps = p0 + p1;
#pragma unroll
            for (int m = 1; m < 16; m <<= 1) ps += __shfl_xor(ps, m, 32);
            srun[r] = srun[r] * corr[r] + ps;
            const int rl = r + (half << 3);          // C-layout row for this lane
            ldsP[wave][rl * 32 + (lane & 15)]      = f2bf(p0);
            ldsP[wave][rl * 32 + 16 + (lane & 15)] = f2bf(p1);
        }
        __syncthreads();   // P visible; K-tile reads complete

        const v16bf pf = load_a_frag(&ldsP[wave][(lane & 15) * 32], half);
#pragma unroll
        for (int nt = 0; nt < 4; ++nt) {
#pragma unroll
            for (int r = 0; r < 8; ++r) oacc[nt][r] *= corr[r];
            v16bf vf = load_b_frag_kmaj(lsV, ALD, nt * 16, lane);
            oacc[nt] = __builtin_amdgcn_wmma_f32_16x16x32_bf16(
                false, pf, false, vf, (short)0, oacc[nt], false, false);
        }
        __syncthreads();   // V-tile/P reads complete before next staging
    }

    // ---- finalize: divide by row sum, write bf16 head slice ----
#pragma unroll
    for (int nt = 0; nt < 4; ++nt) {
#pragma unroll
        for (int r = 0; r < 8; ++r) {
            const int row = q0 + r + (half << 3);
            if (row < CS) {
                const float v = oacc[nt][r] / srun[r];
                O[((long)b * CS + row) * CD + h * CDK + nt * 16 + (lane & 15)] = f2bf(v);
            }
        }
    }
}

// ---------------------------------------------------------------------------
// LayerNorm over D=512; 128 threads per token; writes fp32 + bf16 shadow.
// ---------------------------------------------------------------------------
__global__ void layernorm_k(const float* __restrict__ x,
                            const float* __restrict__ g,
                            const float* __restrict__ bta,
                            float* __restrict__ outF,
                            bf16_t* __restrict__ outB) {
    __shared__ float red[8];
    const int tok = blockIdx.x, tid = threadIdx.x;
    const int lane = tid & 31, wave = tid >> 5;
    const float4 v = ((const float4*)(x + (long)tok * CD))[tid];
    float s  = v.x + v.y + v.z + v.w;
    float sq = v.x * v.x + v.y * v.y + v.z * v.z + v.w * v.w;
#pragma unroll
    for (int m = 16; m >= 1; m >>= 1) {
        s  += __shfl_xor(s, m, 32);
        sq += __shfl_xor(sq, m, 32);
    }
    if (lane == 0) { red[wave * 2] = s; red[wave * 2 + 1] = sq; }
    __syncthreads();
    s  = red[0] + red[2] + red[4] + red[6];
    sq = red[1] + red[3] + red[5] + red[7];
    const float mu  = s * (1.f / CD);
    const float var = sq * (1.f / CD) - mu * mu;
    const float rs  = rsqrtf(var + 1e-5f);
    const float e[4] = { v.x, v.y, v.z, v.w };
#pragma unroll
    for (int j = 0; j < 4; ++j) {
        const int d = tid * 4 + j;
        const float y = (e[j] - mu) * rs * g[d] + bta[d];
        outF[(long)tok * CD + d] = y;
        if (outB) outB[(long)tok * CD + d] = f2bf(y);
    }
}

// ---------------------------------------------------------------------------
// im2col:  vid[B,L,C,H,W] -> patches[B*L*NP, 768] bf16 (pd = ph*48 + pw*3 + c)
// ---------------------------------------------------------------------------
__global__ void im2col_k(const float* __restrict__ vid, bf16_t* __restrict__ patches) {
    const long i = (long)blockIdx.x * 256 + threadIdx.x;
    const long total = (long)CB * CL * CNP * PD;
    if (i >= total) return;
    const int pd = (int)(i % PD);
    long pi = i / PD;
    const int c  = pd % 3;
    const int pw = (pd / 3) % 16;
    const int ph = pd / 48;
    const int wb = (int)(pi % 14); pi /= 14;
    const int hb = (int)(pi % 14); pi /= 14;
    const int l  = (int)(pi % CL);
    const int b  = (int)(pi / CL);
    const int hh = hb * 16 + ph, ww = wb * 16 + pw;
    const float v = vid[((((long)(b * CL + l)) * 3 + c) * 224 + hh) * 224 + ww];
    patches[i] = f2bf(v);
}

// cls/pos assembly -> x fp32 + bf16
__global__ void assemble_k(const float* __restrict__ embed,
                           const float* __restrict__ cls,
                           const float* __restrict__ pos,
                           float* __restrict__ xf, bf16_t* __restrict__ xb) {
    const long i = (long)blockIdx.x * 256 + threadIdx.x;
    const long total = (long)CM * CD;
    if (i >= total) return;
    const int d = (int)(i & (CD - 1));
    long tk = i >> 9;
    const int t = (int)(tk % (CNP + 1)); tk /= (CNP + 1);
    const int l = (int)(tk % CL);
    const int b = (int)(tk / CL);
    float v;
    if (t == 0) v = cls[(long)l * CD + d];
    else        v = embed[(((long)(b * CL + l)) * CNP + (t - 1)) * CD + d];
    v += pos[((long)l * (CNP + 1) + t) * CD + d];
    const long o = (((long)b * CS) + (long)l * (CNP + 1) + t) * CD + d;
    xf[o] = v;
    xb[o] = f2bf(v);
}

__global__ void f32_to_bf16_k(const float* __restrict__ in, bf16_t* __restrict__ out, long n) {
    const long i = (long)blockIdx.x * 256 + threadIdx.x;
    if (i < n) out[i] = f2bf(in[i]);
}

// ---------------------------------------------------------------------------
extern "C" void kernel_launch(void* const* d_in, const int* in_sizes, int n_in,
                              void* d_out, int out_size, void* d_ws, size_t ws_size,
                              hipStream_t stream) {
    (void)in_sizes; (void)n_in; (void)out_size; (void)ws_size;

    const float* vid     = (const float*)d_in[0];
    const float* W_embed = (const float*)d_in[1];
    const float* b_embed = (const float*)d_in[2];
    const float* cls     = (const float*)d_in[3];
    const float* pos     = (const float*)d_in[4];
    const float* Wq = (const float*)d_in[5];  const float* bq = (const float*)d_in[6];
    const float* Wk = (const float*)d_in[7];  const float* bk = (const float*)d_in[8];
    const float* Wv = (const float*)d_in[9];  const float* bv = (const float*)d_in[10];
    const float* Wo = (const float*)d_in[11]; const float* bo = (const float*)d_in[12];
    const float* ln1_g = (const float*)d_in[13]; const float* ln1_b = (const float*)d_in[14];
    const float* W1 = (const float*)d_in[15]; const float* b1 = (const float*)d_in[16];
    const float* W2 = (const float*)d_in[17]; const float* b2 = (const float*)d_in[18];
    const float* ln2_g = (const float*)d_in[19]; const float* ln2_b = (const float*)d_in[20];

    // ---- workspace carve-out (~168 MB) ----
    char* ws = (char*)d_ws;
    size_t off = 0;
    auto alloc = [&](size_t bytes) -> void* {
        void* p = ws + off;
        off += (bytes + 255) & ~(size_t)255;
        return p;
    };
    bf16_t* wemb_bf = (bf16_t*)alloc((size_t)CD * PD * 2);
    bf16_t* wq_bf   = (bf16_t*)alloc((size_t)CNL * CD * CD * 2);
    bf16_t* wk_bf   = (bf16_t*)alloc((size_t)CNL * CD * CD * 2);
    bf16_t* wv_bf   = (bf16_t*)alloc((size_t)CNL * CD * CD * 2);
    bf16_t* wo_bf   = (bf16_t*)alloc((size_t)CNL * CD * CD * 2);
    bf16_t* w1_bf   = (bf16_t*)alloc((size_t)CNL * CFF * CD * 2);
    bf16_t* w2_bf   = (bf16_t*)alloc((size_t)CNL * CD * CFF * 2);
    bf16_t* patches = (bf16_t*)alloc((size_t)CB * CL * CNP * PD * 2);
    float*  embedF  = (float*) alloc((size_t)CB * CL * CNP * CD * 4);
    float*  x_f     = (float*) alloc((size_t)CM * CD * 4);
    bf16_t* x_b     = (bf16_t*)alloc((size_t)CM * CD * 2);
    float*  x1_f    = (float*) alloc((size_t)CM * CD * 4);
    bf16_t* x1_b    = (bf16_t*)alloc((size_t)CM * CD * 2);
    float*  y_f     = (float*) alloc((size_t)CM * CD * 4);
    bf16_t* q_b     = (bf16_t*)alloc((size_t)CM * CD * 2);
    bf16_t* k_b     = (bf16_t*)alloc((size_t)CM * CD * 2);
    bf16_t* v_b     = (bf16_t*)alloc((size_t)CM * CD * 2);
    bf16_t* o_b     = (bf16_t*)alloc((size_t)CM * CD * 2);
    bf16_t* h_b     = (bf16_t*)alloc((size_t)CM * CFF * 2);

    auto cvt = [&](const float* src, bf16_t* dst, long n) {
        f32_to_bf16_k<<<dim3((unsigned)((n + 255) / 256)), dim3(256), 0, stream>>>(src, dst, n);
    };
    cvt(W_embed, wemb_bf, (long)CD * PD);
    cvt(Wq, wq_bf, (long)CNL * CD * CD);
    cvt(Wk, wk_bf, (long)CNL * CD * CD);
    cvt(Wv, wv_bf, (long)CNL * CD * CD);
    cvt(Wo, wo_bf, (long)CNL * CD * CD);
    cvt(W1, w1_bf, (long)CNL * CFF * CD);
    cvt(W2, w2_bf, (long)CNL * CD * CFF);

    auto gemm = [&](const bf16_t* A, const bf16_t* W, const float* bias,
                    const float* res, float* oF, bf16_t* oB,
                    int M, int N, int K, int relu) {
        dim3 grid((M + 127) / 128, N / 64);
        gemm_bf16_wmma<<<grid, dim3(256), 0, stream>>>(A, W, bias, res, oF, oB, M, N, K, relu);
    };

    // ---- patch embedding ----
    {
        const long nIm = (long)CB * CL * CNP * PD;
        im2col_k<<<dim3((unsigned)((nIm + 255) / 256)), dim3(256), 0, stream>>>(vid, patches);
        gemm(patches, wemb_bf, b_embed, nullptr, embedF, nullptr, CB * CL * CNP, CD, PD, 0);
        const long nAs = (long)CM * CD;
        assemble_k<<<dim3((unsigned)((nAs + 255) / 256)), dim3(256), 0, stream>>>(
            embedF, cls, pos, x_f, x_b);
    }

    // ---- encoder layers ----
    for (int l = 0; l < CNL; ++l) {
        const bf16_t* wqL = wq_bf + (size_t)l * CD * CD;
        const bf16_t* wkL = wk_bf + (size_t)l * CD * CD;
        const bf16_t* wvL = wv_bf + (size_t)l * CD * CD;
        const bf16_t* woL = wo_bf + (size_t)l * CD * CD;
        const bf16_t* w1L = w1_bf + (size_t)l * CFF * CD;
        const bf16_t* w2L = w2_bf + (size_t)l * CD * CFF;

        gemm(x_b, wqL, bq + (size_t)l * CD, nullptr, nullptr, q_b, CM, CD, CD, 0);
        gemm(x_b, wkL, bk + (size_t)l * CD, nullptr, nullptr, k_b, CM, CD, CD, 0);
        gemm(x_b, wvL, bv + (size_t)l * CD, nullptr, nullptr, v_b, CM, CD, CD, 0);

        attention_wmma<<<dim3((CS + 63) / 64, CNH, CB), dim3(128), 0, stream>>>(
            q_b, k_b, v_b, o_b);

        gemm(o_b, woL, bo + (size_t)l * CD, x_f, y_f, nullptr, CM, CD, CD, 0);
        layernorm_k<<<dim3(CM), dim3(128), 0, stream>>>(
            y_f, ln1_g + (size_t)l * CD, ln1_b + (size_t)l * CD, x1_f, x1_b);

        gemm(x1_b, w1L, b1 + (size_t)l * CFF, nullptr, nullptr, h_b, CM, CFF, CD, 1);
        gemm(h_b, w2L, b2 + (size_t)l * CD, x1_f, y_f, nullptr, CM, CD, CFF, 0);

        float* xoutF = (l == CNL - 1) ? (float*)d_out : x_f;
        layernorm_k<<<dim3(CM), dim3(128), 0, stream>>>(
            y_f, ln2_g + (size_t)l * CD, ln2_b + (size_t)l * CD, xoutF, x_b);
    }
}
